// WatershedGroups_2851858284930
// MI455X (gfx1250) — matrix operand order repaired and checked by reading, not compiled, not run on hardware
//
#include <hip/hip_runtime.h>

typedef __attribute__((ext_vector_type(16))) _Float16 v16h;
typedef __attribute__((ext_vector_type(8)))  float    v8f;

#define IMG_H 224
#define IMG_W 224
#define NPIX (IMG_H * IMG_W)          // 50176
#define NTHREADS 512                  // 16 wave32 per workgroup (one WGP)
#define PPT (NPIX / NTHREADS)         // 98 pixels per thread, exact
#define ROWW 8                        // u32 words per bitmask row (224 bits -> 7, pad to 8)
#define MIN_DIST 20
#define MAX_SEGS 64
#define CC_ITERS 24
#define FLOOD_ITERS 192
#define MAX_MARKERS 2048
#define QSCALE 4.0f                   // cost quantization: D fits 20 bits (192*1307 < 2^18)
#define DINF 0xFFFFFu
#define PIX_PER_WMMA 160              // 10 groups x 16 pixels packed into K=32
#define NWMMA_BLKS ((NPIX + PIX_PER_WMMA - 1) / PIX_PER_WMMA)   // 314

// LDS layout (bytes): total 323712 <= 320 KB (327680) per workgroup on CDNA5
#define OFF_A    0                    // u32[50176]  : dist (f32 view) -> packed (D<<12|label)
#define OFF_B    200704               // u16[50176]  : row-dist -> CC labels -> markers -> cost_q
#define OFF_FG   301056               // u32[224*8]  : foreground bitmask
#define OFF_PK   308224               // u32[224*8]  : peak bitmask
#define OFF_ROOT 315392               // u32[2048]   : root label list (for rank relabel)
#define OFF_MISC 323584               // u32[32]     : maxdist bits, root count, maxlabel
#define SMEM_BYTES 323712

static __device__ __forceinline__ unsigned umin2(unsigned a, unsigned b) { return a < b ? a : b; }
static __device__ __forceinline__ unsigned umax2(unsigned a, unsigned b) { return a > b ? a : b; }

__global__ __launch_bounds__(NTHREADS, 1)
void watershed_kernel(const float* __restrict__ x, int* __restrict__ out)
{
    extern __shared__ unsigned char smem[];
    unsigned int*   P     = (unsigned int*)  (smem + OFF_A);
    float*          distf = (float*)         (smem + OFF_A);   // aliases P (phase-disjoint)
    unsigned short* Bq    = (unsigned short*)(smem + OFF_B);
    unsigned int*   fgm   = (unsigned int*)  (smem + OFF_FG);
    unsigned int*   pkm   = (unsigned int*)  (smem + OFF_PK);
    unsigned int*   roots = (unsigned int*)  (smem + OFF_ROOT);
    unsigned int*   misc  = (unsigned int*)  (smem + OFF_MISC);

    const int tid = threadIdx.x;
    const int img = blockIdx.x;
    const float* ximg = x + (size_t)img * 3 * NPIX;

    auto fgbit = [&](int h, int w) -> unsigned {
        return (fgm[h * ROWW + (w >> 5)] >> (w & 31)) & 1u;
    };
    auto pkbit = [&](int h, int w) -> unsigned {
        return (pkm[h * ROWW + (w >> 5)] >> (w & 31)) & 1u;
    };

    // ---- phase 0: clear masks & counters -----------------------------------
    for (int i = tid; i < IMG_H * ROWW; i += NTHREADS) { fgm[i] = 0u; pkm[i] = 0u; }
    if (tid < 32) misc[tid] = 0u;
    __syncthreads();

    // ---- phase 1: gray via WMMA, 160 pixels per v_wmma_f32_16x16x32_f16 ----
    // K=32 packs 10 pixel-groups of 16 (3 channels each). B[k][n]=85 iff
    // 3n<=k<=3n+2 (n<10), so D[m][n] = 85*sum_c x_c(group n, pixel m) = gray*?
    // A f16 16x32 layout: lanes 0-15 (M=lane): elems 0-7 = K0-7, 8-15 = K16-23;
    //                     lanes 16-31 (M=lane-16): elems = K8-15, K24-31.
    // B f16 32x16 layout: lanes 0-15 (N=lane): elems 0-7 = K0-7, 8-15 = K16-23;
    //                     lanes 16-31 (N=lane-16): elems = K8-15, K24-31.
    // D f32 16x16 layout: lane<16: c[r] = D[r][lane]; lane>=16: c[r] = D[8+r][lane-16].
    {
        const int wave = tid >> 5, lane = tid & 31;
        const bool hi  = lane >= 16;
        const int  m   = lane & 15;      // A row for this lane
        const int  n   = lane & 15;      // B/D column for this lane

        v16h b;                          // loop-invariant constant weights
        #pragma unroll
        for (int e = 0; e < 16; ++e) {
            const int K = ((e < 8) ? e : (e + 8)) + (hi ? 8 : 0);
            const bool on = (n < 10) && (K >= 3 * n) && (K <= 3 * n + 2);
            b[e] = on ? (_Float16)85.0f : (_Float16)0.0f;
        }

        for (int gb = wave; gb < NWMMA_BLKS; gb += (NTHREADS >> 5)) {
            const int base = gb * PIX_PER_WMMA;
            v16h a;
            #pragma unroll
            for (int e = 0; e < 16; ++e) {
                int K = (e < 8) ? e : (e + 8);
                if (hi) K += 8;
                const int g  = K / 3, ch = K - 3 * g;     // g<10 when K<30
                const int p  = base + g * 16 + m;
                const bool valid = (K < 30) && (p < NPIX);
                const int pc = valid ? p : 0;
                const float v = ximg[ch * NPIX + pc];
                a[e] = valid ? (_Float16)v : (_Float16)0.0f;
            }
            v8f c = {};
            c = __builtin_amdgcn_wmma_f32_16x16x32_f16(false, a, false, b,
                                                       (short)0, c, false, false);
            // lane n<10 owns pixels base+16n+0..7 (hi: +8..15): constant-index extract
            if (n < 10) {
                const int pb = base + n * 16 + (hi ? 8 : 0);   // 8-aligned
                unsigned mbits = 0u;
                #pragma unroll
                for (int r = 0; r < 8; ++r)
                    mbits |= (c[r] >= 1.0f) ? (1u << r) : 0u;  // floor(clip(mean*255)) > 0
                if (mbits && pb < NPIX) {
                    const int h = pb / IMG_W, w = pb % IMG_W;  // pb..pb+7 same mask word
                    atomicOr(&fgm[h * ROWW + (w >> 5)], mbits << (w & 31));
                }
            }
        }
    }
    __syncthreads();

    // ---- phase 2: EDT pass 1 (per-column nearest-bg row distance, u16) -----
    for (int w = tid; w < IMG_W; w += NTHREADS) {
        unsigned run = 30000u;
        for (int h = 0; h < IMG_H; ++h) {
            run = fgbit(h, w) ? umin2(run + 1u, 30000u) : 0u;
            Bq[h * IMG_W + w] = (unsigned short)run;
        }
        run = 30000u;
        for (int h = IMG_H - 1; h >= 0; --h) {
            run = fgbit(h, w) ? umin2(run + 1u, 30000u) : 0u;
            if (run < (unsigned)Bq[h * IMG_W + w]) Bq[h * IMG_W + w] = (unsigned short)run;
        }
    }
    __syncthreads();

    // ---- phase 3: EDT pass 2 (min-plus envelope along rows), 8-wide blocked -
    for (int t = tid; t < IMG_H * (IMG_W / 8); t += NTHREADS) {
        const int h  = t / (IMG_W / 8);
        const int wb = (t % (IMG_W / 8)) * 8;
        unsigned m[8];
        #pragma unroll
        for (int j = 0; j < 8; ++j) m[j] = 0xFFFFFFFFu;
        const unsigned short* row = Bq + h * IMG_W;
        for (int k = 0; k < IMG_W; ++k) {
            const unsigned rd   = row[k];
            const unsigned base = rd * rd;
            #pragma unroll
            for (int j = 0; j < 8; ++j) {
                const int dw = (wb + j) - k;
                m[j] = umin2(m[j], base + (unsigned)(dw * dw));
            }
        }
        #pragma unroll
        for (int j = 0; j < 8; ++j) distf[h * IMG_W + wb + j] = sqrtf((float)m[j]);
    }
    __syncthreads();

    // ---- phase 4: max(dist) reduction + 10x10 peak filter ------------------
    {
        float lmax = 0.0f;
        for (int i = 0; i < PPT; ++i) lmax = fmaxf(lmax, distf[tid + i * NTHREADS]);
        atomicMax(&misc[0], __float_as_uint(lmax));  // dist >= 0: u32 order == f32 order
        for (int i = 0; i < PPT; ++i) {
            const int p = tid + i * NTHREADS;
            const int h = p / IMG_W, w = p % IMG_W;
            if (h < MIN_DIST || h >= IMG_H - MIN_DIST ||
                w < MIN_DIST || w >= IMG_W - MIN_DIST) continue;   // border exclusion
            if (!fgbit(h, w)) continue;
            const float d0 = distf[p];
            if (!(d0 > 0.0f)) continue;
            float mx = -1.0f;   // SAME padding for window 10: rows [h-4,h+5] (in-range here)
            for (int dy = -4; dy <= 5; ++dy)
                for (int dx = -4; dx <= 5; ++dx)
                    mx = fmaxf(mx, distf[(h + dy) * IMG_W + (w + dx)]);
            if (d0 >= mx) atomicOr(&pkm[h * ROWW + (w >> 5)], 1u << (w & 31));
        }
    }
    __syncthreads();

    // ---- phase 5: 4-conn CC on peaks via monotone max-label propagation ----
    for (int i = 0; i < PPT; ++i) {
        const int p = tid + i * NTHREADS;
        const int h = p / IMG_W, w = p % IMG_W;
        Bq[p] = pkbit(h, w) ? (unsigned short)(p + 1) : (unsigned short)0;
    }
    __syncthreads();
    for (int it = 0; it < CC_ITERS; ++it) {
        for (int i = 0; i < PPT; ++i) {
            const int p = tid + i * NTHREADS;
            const int h = p / IMG_W, w = p % IMG_W;
            if (!pkbit(h, w)) continue;           // peaks are interior: p+-1, p+-W valid
            unsigned l = Bq[p];
            l = umax2(l, (unsigned)Bq[p - IMG_W]);
            l = umax2(l, (unsigned)Bq[p + IMG_W]);
            l = umax2(l, (unsigned)Bq[p - 1]);
            l = umax2(l, (unsigned)Bq[p + 1]);
            Bq[p] = (unsigned short)l;
        }
        __syncthreads();
    }

    // ---- phase 6: collect root labels (plateau-max pixels) -----------------
    for (int i = 0; i < PPT; ++i) {
        const int p = tid + i * NTHREADS;
        const int h = p / IMG_W, w = p % IMG_W;
        if (pkbit(h, w) && Bq[p] == (unsigned short)(p + 1)) {
            const unsigned idx = atomicAdd(&misc[1], 1u);
            if (idx < MAX_MARKERS) roots[idx] = (unsigned)(p + 1);
        }
    }
    __syncthreads();

    // ---- phase 7: rank relabel (== unique + searchsorted) ------------------
    {
        const unsigned n = umin2(misc[1], (unsigned)MAX_MARKERS);
        for (int i = 0; i < PPT; ++i) {
            const int p = tid + i * NTHREADS;
            const int h = p / IMG_W, w = p % IMG_W;
            if (!pkbit(h, w)) continue;
            const unsigned lbl = Bq[p];
            unsigned r = 1u;
            for (unsigned k = 0; k < n; ++k) r += (roots[k] < lbl) ? 1u : 0u;
            Bq[p] = (unsigned short)r;          // markers: 1..n
        }
    }
    __syncthreads();

    // ---- phase 8: build packed flood state + quantized cost ----------------
    // P[p] = (D<<12)|label with D in 20 bits; Bq[p] = cost_q (u16).
    {
        const float maxd = __uint_as_float(misc[0]);
        for (int i = 0; i < PPT; ++i) {
            const int p = tid + i * NTHREADS;
            const float d = distf[p];                         // read before overwrite (same thread)
            const unsigned m = Bq[p];
            const unsigned short cq =
                (unsigned short)(((maxd - d) + 10.0f) * QSCALE + 0.5f);
            P[p]  = (m > 0u) ? m : (DINF << 12);              // seeds at D=0, else INF
            Bq[p] = cq;
        }
    }
    __syncthreads();

    // ---- phase 9: compact watershed flood: packed ds_min_u32 relaxation ----
    for (int it = 0; it < FLOOD_ITERS; ++it) {
        for (int i = 0; i < PPT; ++i) {
            const int p = tid + i * NTHREADS;
            const int h = p / IMG_W, w = p % IMG_W;
            if (!fgbit(h, w)) continue;
            const unsigned cur = P[p];
            if ((cur >> 12) == 0u) continue;                  // seed: cannot improve
            const unsigned cq = Bq[p];
            unsigned best = cur;
            if (h > 0) {
                const unsigned q = P[p - IMG_W], nd = (q >> 12) + cq;
                if (nd < DINF) best = umin2(best, (nd << 12) | (q & 0xFFFu));
            }
            if (h < IMG_H - 1) {
                const unsigned q = P[p + IMG_W], nd = (q >> 12) + cq;
                if (nd < DINF) best = umin2(best, (nd << 12) | (q & 0xFFFu));
            }
            if (w > 0) {
                const unsigned q = P[p - 1], nd = (q >> 12) + cq;
                if (nd < DINF) best = umin2(best, (nd << 12) | (q & 0xFFFu));
            }
            if (w < IMG_W - 1) {
                const unsigned q = P[p + 1], nd = (q >> 12) + cq;
                if (nd < DINF) best = umin2(best, (nd << 12) | (q & 0xFFFu));
            }
            if (best < cur) atomicMin(&P[p], best);           // ds_min_u32: monotone, race-safe
        }
        __syncthreads();
    }

    // ---- phase 10: max label, divide, write int32 output -------------------
    {
        unsigned lmax = 0u;
        for (int i = 0; i < PPT; ++i) lmax = umax2(lmax, P[tid + i * NTHREADS] & 0xFFFu);
        atomicMax(&misc[2], lmax);
    }
    __syncthreads();
    {
        const unsigned div = misc[2] / MAX_SEGS + 1u;
        int* o = out + (size_t)img * NPIX;
        for (int i = 0; i < PPT; ++i) {
            const int p = tid + i * NTHREADS;
            o[p] = (int)((P[p] & 0xFFFu) / div);
        }
    }
}

extern "C" void kernel_launch(void* const* d_in, const int* in_sizes, int n_in,
                              void* d_out, int out_size, void* d_ws, size_t ws_size,
                              hipStream_t stream) {
    const float* x = (const float*)d_in[0];
    int* out = (int*)d_out;
    const int nimg = in_sizes[0] / (3 * NPIX);   // B = 16

    // Opt in to >64KB dynamic LDS (CDNA5 WGP: up to 320KB per workgroup).
    (void)hipFuncSetAttribute((const void*)watershed_kernel,
                              hipFuncAttributeMaxDynamicSharedMemorySize, SMEM_BYTES);

    hipLaunchKernelGGL(watershed_kernel, dim3(nimg), dim3(NTHREADS), SMEM_BYTES, stream,
                       x, out);
}
